// ChannelAttention_53996328845652
// MI455X (gfx1250) — matrix operand order repaired and checked by reading, not compiled, not run on hardware
//
#include <hip/hip_runtime.h>

// ---------------------------------------------------------------------------
// ChannelAttention on MI455X (gfx1250, wave32, WMMA)
//   B=8, N=4096, C=512
//   out = ( relu(x@Wqkv+b) -> q,k,v ; softmax(q^T k /sqrt(C)) * adj ; v@attn ) @ Wo + bo
// All four GEMMs: v_wmma_f32_16x16x32_f16, 8 waves/block, 64x128 block tile,
// 32x32 per-wave tile (4 WMMA / wave / K-step), double-buffered LDS,
// one workgroup barrier per K-step.
// ---------------------------------------------------------------------------

typedef __attribute__((ext_vector_type(16))) _Float16 v16h;
typedef __attribute__((ext_vector_type(8)))  float    v8f;

#define BB   8
#define NN_  4096
#define CC   512
#define C3   1536
#define LDSK 40          // padded LDS row stride in f16 (80 B, 16 B aligned)

// ---- WMMA fragment loads from LDS tiles -----------------------------------
// A tile stored row-major  As[m][k]  (16 rows x 32 k per fragment)
// B tile stored col-major  Bs[n][k]  (16 cols x 32 k per fragment)

__device__ __forceinline__ v16h frag_a(const _Float16* base, int lane) {
  const int m = lane & 15, sel = lane >> 4;            // lanes>=16 hold K {8..15,24..31}
  const _Float16* row = base + m * LDSK + sel * 8;
  union { v16h v; uint4 q[2]; } u;
  u.q[0] = *reinterpret_cast<const uint4*>(row);       // K = sel*8 .. +7
  u.q[1] = *reinterpret_cast<const uint4*>(row + 16);  // K = sel*8+16 .. +23
  return u.v;
}

__device__ __forceinline__ v16h frag_b(const _Float16* base, int lane) {
  const int n = lane & 15, sel = lane >> 4;            // lanes>=16 hold K {8..15,24..31}
  const _Float16* p = base + n * LDSK + sel * 8;
  union { v16h v; uint4 q[2]; } u;
  u.q[0] = *reinterpret_cast<const uint4*>(p);
  u.q[1] = *reinterpret_cast<const uint4*>(p + 16);
  return u.v;
}

__device__ __forceinline__ v8f wmma_step(v16h a, v16h b, v8f c) {
  return __builtin_amdgcn_wmma_f32_16x16x32_f16(false, a, false, b, (short)0, c,
                                                false, false);
}

__device__ __forceinline__ void cvt8(const float4& f0, const float4& f1, uint4& out) {
  union { _Float16 h[8]; uint4 q; } u;
  u.h[0]=(_Float16)f0.x; u.h[1]=(_Float16)f0.y; u.h[2]=(_Float16)f0.z; u.h[3]=(_Float16)f0.w;
  u.h[4]=(_Float16)f1.x; u.h[5]=(_Float16)f1.y; u.h[6]=(_Float16)f1.z; u.h[7]=(_Float16)f1.w;
  out = u.q;
}

// ---------------------------------------------------------------------------
// Kernel 1: qkv = relu(x @ Wqkv + bqkv)  -> f16   [32768 x 1536]
// ---------------------------------------------------------------------------
__global__ void __launch_bounds__(256)
qkv_kernel(const float* __restrict__ x, const float* __restrict__ Wqkv,
           const float* __restrict__ bqkv, _Float16* __restrict__ qkvh) {
  __shared__ _Float16 As[2][64 * LDSK];    // [m][k]
  __shared__ _Float16 Bs[2][128 * LDSK];   // [n][k]
  const int t = threadIdx.x, lane = t & 31, wave = t >> 5;
  const int m0 = blockIdx.x * 64, n0 = blockIdx.y * 128;
  const int wm = (wave & 1) * 32, wn = (wave >> 1) * 32;

  const int am = t >> 2, akg = (t & 3) * 8;        // A: 64 rows x 32 k, 8/thread
  const int bk = t & 31, bng = (t >> 5) * 16;      // B: 128 n x 32 k, 16/thread

  float4 a0, a1, b0, b1, b2, b3;
  auto loadA = [&](int kk) {
    const float* src = x + (size_t)(m0 + am) * CC + kk + akg;
    a0 = reinterpret_cast<const float4*>(src)[0];
    a1 = reinterpret_cast<const float4*>(src)[1];
  };
  auto loadB = [&](int kk) {
    const float* src = Wqkv + (size_t)(kk + bk) * C3 + n0 + bng;
    b0 = reinterpret_cast<const float4*>(src)[0];
    b1 = reinterpret_cast<const float4*>(src)[1];
    b2 = reinterpret_cast<const float4*>(src)[2];
    b3 = reinterpret_cast<const float4*>(src)[3];
  };
  auto stage = [&](int buf) {
    uint4 q; cvt8(a0, a1, q);
    *reinterpret_cast<uint4*>(&As[buf][am * LDSK + akg]) = q;
    const float f[16] = {b0.x,b0.y,b0.z,b0.w, b1.x,b1.y,b1.z,b1.w,
                         b2.x,b2.y,b2.z,b2.w, b3.x,b3.y,b3.z,b3.w};
#pragma unroll
    for (int j = 0; j < 16; ++j) Bs[buf][(bng + j) * LDSK + bk] = (_Float16)f[j];
  };

  v8f acc00 = {}, acc01 = {}, acc10 = {}, acc11 = {};
  loadA(0); loadB(0); stage(0);
  const int KSTEPS = CC / 32;
  for (int s = 0; s < KSTEPS; ++s) {
    __syncthreads();
    const int cur = s & 1;
    if (s + 1 < KSTEPS) { loadA((s + 1) * 32); loadB((s + 1) * 32); }
    v16h af0 = frag_a(&As[cur][wm * LDSK], lane);
    v16h af1 = frag_a(&As[cur][(wm + 16) * LDSK], lane);
    v16h bf0 = frag_b(&Bs[cur][wn * LDSK], lane);
    v16h bf1 = frag_b(&Bs[cur][(wn + 16) * LDSK], lane);
    acc00 = wmma_step(af0, bf0, acc00);
    acc01 = wmma_step(af0, bf1, acc01);
    acc10 = wmma_step(af1, bf0, acc10);
    acc11 = wmma_step(af1, bf1, acc11);
    if (s + 1 < KSTEPS) stage(1 - cur);
  }

  const int nl = lane & 15, mh = (lane >> 4) * 8;
#pragma unroll
  for (int ni = 0; ni < 2; ++ni) {
    const int n = n0 + wn + ni * 16 + nl;
    const float bias = bqkv[n];
#pragma unroll
    for (int mi = 0; mi < 2; ++mi) {
      const v8f& A = (mi == 0) ? (ni == 0 ? acc00 : acc01)
                               : (ni == 0 ? acc10 : acc11);
#pragma unroll
      for (int r = 0; r < 8; ++r) {
        const int m = m0 + wm + mi * 16 + mh + r;
        float v = A[r] + bias;
        v = v > 0.f ? v : 0.f;
        qkvh[(size_t)m * C3 + n] = (_Float16)v;
      }
    }
  }
}

// ---------------------------------------------------------------------------
// Kernel 2: scores[b,c,d] = (1/sqrt(C)) * sum_n q[b,n,c]*k[b,n,d]   (fp32 out)
// ---------------------------------------------------------------------------
__global__ void __launch_bounds__(256)
scores_kernel(const _Float16* __restrict__ qkvh, float* __restrict__ attn) {
  __shared__ _Float16 As[2][64 * LDSK];    // As[c][n]
  __shared__ _Float16 Bs[2][128 * LDSK];   // Bs[d][n]
  const int t = threadIdx.x, lane = t & 31, wave = t >> 5;
  const int c0 = blockIdx.x * 64, d0 = blockIdx.y * 128, b = blockIdx.z;
  const size_t bN = (size_t)b * NN_;
  const int wm = (wave & 1) * 32, wn = (wave >> 1) * 32;

  const int an = t & 31, acg = (t >> 5) * 8;       // A: 64 c x 32 n, 8/thread
  const int bn = t & 31, bdg = (t >> 5) * 16;      // B: 128 d x 32 n, 16/thread

  uint4 ra, rb0, rb1;
  auto loadA = [&](int nn) {
    ra = *reinterpret_cast<const uint4*>(qkvh + (bN + nn + an) * C3 + c0 + acg);
  };
  auto loadB = [&](int nn) {
    const _Float16* src = qkvh + (bN + nn + bn) * C3 + CC + d0 + bdg;  // k block
    rb0 = reinterpret_cast<const uint4*>(src)[0];
    rb1 = reinterpret_cast<const uint4*>(src)[1];
  };
  auto stage = [&](int buf) {
    const _Float16* ha = reinterpret_cast<const _Float16*>(&ra);
#pragma unroll
    for (int j = 0; j < 8; ++j) As[buf][(acg + j) * LDSK + an] = ha[j];
    const _Float16* hb0 = reinterpret_cast<const _Float16*>(&rb0);
    const _Float16* hb1 = reinterpret_cast<const _Float16*>(&rb1);
#pragma unroll
    for (int j = 0; j < 8; ++j) Bs[buf][(bdg + j) * LDSK + bn] = hb0[j];
#pragma unroll
    for (int j = 0; j < 8; ++j) Bs[buf][(bdg + 8 + j) * LDSK + bn] = hb1[j];
  };

  v8f acc00 = {}, acc01 = {}, acc10 = {}, acc11 = {};
  loadA(0); loadB(0); stage(0);
  const int KSTEPS = NN_ / 32;
  for (int s = 0; s < KSTEPS; ++s) {
    __syncthreads();
    const int cur = s & 1;
    if (s + 1 < KSTEPS) { loadA((s + 1) * 32); loadB((s + 1) * 32); }
    v16h af0 = frag_a(&As[cur][wm * LDSK], lane);
    v16h af1 = frag_a(&As[cur][(wm + 16) * LDSK], lane);
    v16h bf0 = frag_b(&Bs[cur][wn * LDSK], lane);
    v16h bf1 = frag_b(&Bs[cur][(wn + 16) * LDSK], lane);
    acc00 = wmma_step(af0, bf0, acc00);
    acc01 = wmma_step(af0, bf1, acc01);
    acc10 = wmma_step(af1, bf0, acc10);
    acc11 = wmma_step(af1, bf1, acc11);
    if (s + 1 < KSTEPS) stage(1 - cur);
  }

  const float scale = 0.044194173824159216f;       // 1/sqrt(512)
  const int nl = lane & 15, mh = (lane >> 4) * 8;
#pragma unroll
  for (int ni = 0; ni < 2; ++ni) {
    const int d = d0 + wn + ni * 16 + nl;
#pragma unroll
    for (int mi = 0; mi < 2; ++mi) {
      const v8f& A = (mi == 0) ? (ni == 0 ? acc00 : acc01)
                               : (ni == 0 ? acc10 : acc11);
#pragma unroll
      for (int r = 0; r < 8; ++r) {
        const int c = c0 + wm + mi * 16 + mh + r;
        attn[((size_t)b * CC + c) * CC + d] = A[r] * scale;
      }
    }
  }
}

// ---------------------------------------------------------------------------
// Kernel 3: row softmax over d, then elementwise * adj   (in place, fp32)
// ---------------------------------------------------------------------------
__global__ void __launch_bounds__(128)
softmax_adj_kernel(float* __restrict__ attn, const float* __restrict__ adj) {
  const int row = blockIdx.x;                      // b*512 + c, 4096 rows
  float* p = attn + (size_t)row * CC;
  const float* g = adj + (size_t)row * CC;
  const int t = threadIdx.x;
  __shared__ float red[128];

  float4 v = reinterpret_cast<float4*>(p)[t];      // 128*4 = 512
  float m = fmaxf(fmaxf(v.x, v.y), fmaxf(v.z, v.w));
  red[t] = m; __syncthreads();
  for (int s = 64; s > 0; s >>= 1) {
    if (t < s) red[t] = fmaxf(red[t], red[t + s]);
    __syncthreads();
  }
  const float M = red[0];
  __syncthreads();

  float4 e;
  e.x = __expf(v.x - M); e.y = __expf(v.y - M);
  e.z = __expf(v.z - M); e.w = __expf(v.w - M);
  red[t] = e.x + e.y + e.z + e.w; __syncthreads();
  for (int s = 64; s > 0; s >>= 1) {
    if (t < s) red[t] += red[t + s];
    __syncthreads();
  }
  const float inv = 1.f / red[0];

  float4 a4 = reinterpret_cast<const float4*>(g)[t];
  float4 o;
  o.x = e.x * inv * a4.x; o.y = e.y * inv * a4.y;
  o.z = e.z * inv * a4.z; o.w = e.w * inv * a4.w;
  reinterpret_cast<float4*>(p)[t] = o;
}

// ---------------------------------------------------------------------------
// Kernel 4: y[b,n,d] = sum_c v[b,n,c] * attn[b,c,d]   -> f16
// ---------------------------------------------------------------------------
__global__ void __launch_bounds__(256)
av_kernel(const _Float16* __restrict__ qkvh, const float* __restrict__ attn,
          _Float16* __restrict__ yh) {
  __shared__ _Float16 As[2][64 * LDSK];    // As[n][c]
  __shared__ _Float16 Bs[2][128 * LDSK];   // Bs[d][c]
  const int t = threadIdx.x, lane = t & 31, wave = t >> 5;
  const int m0 = blockIdx.x * 64, n0 = blockIdx.y * 128, b = blockIdx.z;
  const size_t bN = (size_t)b * NN_;
  const int wm = (wave & 1) * 32, wn = (wave >> 1) * 32;

  const int am = t >> 2, akg = (t & 3) * 8;        // A: 64 m x 32 k (f16 rows)
  const int bk = t & 31, bdg = (t >> 5) * 16;      // B: 128 d x 32 k (from fp32)

  uint4 ra;
  float4 b0, b1, b2, b3;
  auto loadA = [&](int kk) {
    ra = *reinterpret_cast<const uint4*>(qkvh + (bN + m0 + am) * C3 + 2 * CC + kk + akg);
  };
  auto loadB = [&](int kk) {
    const float* src = attn + ((size_t)b * CC + kk + bk) * CC + n0 + bdg;
    b0 = reinterpret_cast<const float4*>(src)[0];
    b1 = reinterpret_cast<const float4*>(src)[1];
    b2 = reinterpret_cast<const float4*>(src)[2];
    b3 = reinterpret_cast<const float4*>(src)[3];
  };
  auto stage = [&](int buf) {
    *reinterpret_cast<uint4*>(&As[buf][am * LDSK + akg]) = ra;
    const float f[16] = {b0.x,b0.y,b0.z,b0.w, b1.x,b1.y,b1.z,b1.w,
                         b2.x,b2.y,b2.z,b2.w, b3.x,b3.y,b3.z,b3.w};
#pragma unroll
    for (int j = 0; j < 16; ++j) Bs[buf][(bdg + j) * LDSK + bk] = (_Float16)f[j];
  };

  v8f acc00 = {}, acc01 = {}, acc10 = {}, acc11 = {};
  loadA(0); loadB(0); stage(0);
  const int KSTEPS = CC / 32;
  for (int s = 0; s < KSTEPS; ++s) {
    __syncthreads();
    const int cur = s & 1;
    if (s + 1 < KSTEPS) { loadA((s + 1) * 32); loadB((s + 1) * 32); }
    v16h af0 = frag_a(&As[cur][wm * LDSK], lane);
    v16h af1 = frag_a(&As[cur][(wm + 16) * LDSK], lane);
    v16h bf0 = frag_b(&Bs[cur][wn * LDSK], lane);
    v16h bf1 = frag_b(&Bs[cur][(wn + 16) * LDSK], lane);
    acc00 = wmma_step(af0, bf0, acc00);
    acc01 = wmma_step(af0, bf1, acc01);
    acc10 = wmma_step(af1, bf0, acc10);
    acc11 = wmma_step(af1, bf1, acc11);
    if (s + 1 < KSTEPS) stage(1 - cur);
  }

  const int nl = lane & 15, mh = (lane >> 4) * 8;
#pragma unroll
  for (int ni = 0; ni < 2; ++ni) {
    const int n = n0 + wn + ni * 16 + nl;
#pragma unroll
    for (int mi = 0; mi < 2; ++mi) {
      const v8f& A = (mi == 0) ? (ni == 0 ? acc00 : acc01)
                               : (ni == 0 ? acc10 : acc11);
#pragma unroll
      for (int r = 0; r < 8; ++r) {
        const int m = m0 + wm + mi * 16 + mh + r;
        yh[(bN + m) * CC + n] = (_Float16)A[r];
      }
    }
  }
}

// ---------------------------------------------------------------------------
// Kernel 5: out = y @ Wo + bo   (fp32 out)
// ---------------------------------------------------------------------------
__global__ void __launch_bounds__(256)
out_kernel(const _Float16* __restrict__ yh, const float* __restrict__ Wo,
           const float* __restrict__ bo, float* __restrict__ out) {
  __shared__ _Float16 As[2][64 * LDSK];
  __shared__ _Float16 Bs[2][128 * LDSK];
  const int t = threadIdx.x, lane = t & 31, wave = t >> 5;
  const int m0 = blockIdx.x * 64, n0 = blockIdx.y * 128;
  const int wm = (wave & 1) * 32, wn = (wave >> 1) * 32;

  const int am = t >> 2, akg = (t & 3) * 8;
  const int bk = t & 31, bng = (t >> 5) * 16;

  uint4 ra;
  float4 b0, b1, b2, b3;
  auto loadA = [&](int kk) {
    ra = *reinterpret_cast<const uint4*>(yh + (size_t)(m0 + am) * CC + kk + akg);
  };
  auto loadB = [&](int kk) {
    const float* src = Wo + (size_t)(kk + bk) * CC + n0 + bng;
    b0 = reinterpret_cast<const float4*>(src)[0];
    b1 = reinterpret_cast<const float4*>(src)[1];
    b2 = reinterpret_cast<const float4*>(src)[2];
    b3 = reinterpret_cast<const float4*>(src)[3];
  };
  auto stage = [&](int buf) {
    *reinterpret_cast<uint4*>(&As[buf][am * LDSK + akg]) = ra;
    const float f[16] = {b0.x,b0.y,b0.z,b0.w, b1.x,b1.y,b1.z,b1.w,
                         b2.x,b2.y,b2.z,b2.w, b3.x,b3.y,b3.z,b3.w};
#pragma unroll
    for (int j = 0; j < 16; ++j) Bs[buf][(bng + j) * LDSK + bk] = (_Float16)f[j];
  };

  v8f acc00 = {}, acc01 = {}, acc10 = {}, acc11 = {};
  loadA(0); loadB(0); stage(0);
  const int KSTEPS = CC / 32;
  for (int s = 0; s < KSTEPS; ++s) {
    __syncthreads();
    const int cur = s & 1;
    if (s + 1 < KSTEPS) { loadA((s + 1) * 32); loadB((s + 1) * 32); }
    v16h af0 = frag_a(&As[cur][wm * LDSK], lane);
    v16h af1 = frag_a(&As[cur][(wm + 16) * LDSK], lane);
    v16h bf0 = frag_b(&Bs[cur][wn * LDSK], lane);
    v16h bf1 = frag_b(&Bs[cur][(wn + 16) * LDSK], lane);
    acc00 = wmma_step(af0, bf0, acc00);
    acc01 = wmma_step(af0, bf1, acc01);
    acc10 = wmma_step(af1, bf0, acc10);
    acc11 = wmma_step(af1, bf1, acc11);
    if (s + 1 < KSTEPS) stage(1 - cur);
  }

  const int nl = lane & 15, mh = (lane >> 4) * 8;
#pragma unroll
  for (int ni = 0; ni < 2; ++ni) {
    const int n = n0 + wn + ni * 16 + nl;
    const float bias = bo[n];
#pragma unroll
    for (int mi = 0; mi < 2; ++mi) {
      const v8f& A = (mi == 0) ? (ni == 0 ? acc00 : acc01)
                               : (ni == 0 ? acc10 : acc11);
#pragma unroll
      for (int r = 0; r < 8; ++r) {
        const int m = m0 + wm + mi * 16 + mh + r;
        out[(size_t)m * CC + n] = A[r] + bias;
      }
    }
  }
}

// ---------------------------------------------------------------------------
extern "C" void kernel_launch(void* const* d_in, const int* in_sizes, int n_in,
                              void* d_out, int out_size, void* d_ws, size_t ws_size,
                              hipStream_t stream) {
  (void)in_sizes; (void)n_in; (void)out_size; (void)ws_size;
  const float* x    = (const float*)d_in[0];
  const float* adj  = (const float*)d_in[1];
  const float* Wqkv = (const float*)d_in[2];
  const float* bqkv = (const float*)d_in[3];
  const float* Wo   = (const float*)d_in[4];
  const float* bo   = (const float*)d_in[5];
  float* out = (float*)d_out;

  char* ws = (char*)d_ws;
  const size_t qkv_bytes  = (size_t)BB * NN_ * C3 * sizeof(_Float16); // 96 MB
  const size_t attn_bytes = (size_t)BB * CC * CC * sizeof(float);     //  8 MB
  _Float16* qkvh = (_Float16*)ws;
  float*    attn = (float*)(ws + qkv_bytes);
  _Float16* yh   = (_Float16*)(ws + qkv_bytes + attn_bytes);          // 32 MB

  qkv_kernel        <<<dim3((BB * NN_) / 64, C3 / 128),       256, 0, stream>>>(x, Wqkv, bqkv, qkvh);
  scores_kernel     <<<dim3(CC / 64, CC / 128, BB),           256, 0, stream>>>(qkvh, attn);
  softmax_adj_kernel<<<dim3(BB * CC),                         128, 0, stream>>>(attn, adj);
  av_kernel         <<<dim3(NN_ / 64, CC / 128, BB),          256, 0, stream>>>(qkvh, attn, yh);
  out_kernel        <<<dim3((BB * NN_) / 64, CC / 128),       256, 0, stream>>>(yh, Wo, bo, out);
}
